// MultiHeadedAttention_81801947120046
// MI455X (gfx1250) — matrix operand order repaired and checked by reading, not compiled, not run on hardware
//
#include <hip/hip_runtime.h>
#include <hip/hip_bf16.h>

typedef __attribute__((ext_vector_type(16))) _Float16 v16h;
typedef __attribute__((ext_vector_type(8)))  float    v8f;
typedef __attribute__((address_space(3)))    _Float16 lds_half_t;

#define WMMA_F16(a, b, c) \
    __builtin_amdgcn_wmma_f32_16x16x32_f16(false, (a), false, (b), (short)0, (c), false, false)

// ---------------------------------------------------------------------------
// Fragment loaders (CDNA5 wave32 WMMA layouts, 16x16x32 f16 -> f32)
//
// A (16x32, MxK): lane<16 -> row=lane,   k in {0..7, 16..23}
//                 lane>=16-> row=lane-16,k in {8..15,24..31}
// B (32x16, KxN): lane<16 -> n=lane,   k in {0..7, 16..23}
//                 lane>=16-> n=lane-16,k in {8..15,24..31}
// C/D (16x16):    elem i -> (row = i + 8*(lane>=16), col = lane&15)
// ---------------------------------------------------------------------------
__device__ __forceinline__ v16h load_a_frag(const _Float16* base, int ld, int lane) {
    int grp = (lane >> 4) & 1;
    int r   = lane & 15;
    const _Float16* p = base + (size_t)r * ld + (grp << 3);
    union { uint4 u[2]; v16h h; } u;
    u.u[0] = *(const uint4*)(p);        // k: koff .. koff+7
    u.u[1] = *(const uint4*)(p + 16);   // k: 16+koff .. 16+koff+7
    return u.h;
}

__device__ __forceinline__ v16h load_b_frag(const _Float16* base, int ld, int lane) {
    int grp = (lane >> 4) & 1;
    int r   = lane & 15;
    const _Float16* p = base + (size_t)r * ld + (grp << 3);
    union { uint4 u[2]; v16h h; } u;
    u.u[0] = *(const uint4*)(p);
    u.u[1] = *(const uint4*)(p + 16);
    return u.h;
}

// ---------------------------------------------------------------------------
// CDNA5 async global->LDS copy (16B per lane), tracked by ASYNCcnt
// ---------------------------------------------------------------------------
__device__ __forceinline__ void async_copy_b128(lds_half_t* l, const _Float16* g) {
    asm volatile("global_load_async_to_lds_b128 %0, %1, off"
                 :: "v"(l), "v"(g) : "memory");
}
__device__ __forceinline__ void wait_asynccnt0() {
    asm volatile("s_wait_asynccnt 0" ::: "memory");
}

// ---------------------------------------------------------------------------
// f32 -> f16 conversion (weights, done once)
// ---------------------------------------------------------------------------
__global__ void cvt_f16_kernel(const float* __restrict__ src,
                               _Float16* __restrict__ dst, int n) {
    int i = blockIdx.x * blockDim.x + threadIdx.x;
    if (i < n) dst[i] = (_Float16)src[i];
}

// ---------------------------------------------------------------------------
// Projection kernel: Y[M,512] = X[M,512] @ W[512,512]^T + bias  (W in f16)
// Block: 256 threads = 8 waves. Block tile: 16 rows x 128 cols (16 cols/wave).
// X row-tile (16x32) staged to LDS as f16 once per k-step, shared by 8 waves.
// OUT_TRANS stores V transposed as [B, H, d_k, T] f16 with packed b128 stores.
// ---------------------------------------------------------------------------
template <bool IN_HALF, bool OUT_F32, bool OUT_TRANS>
__global__ void proj_kernel(const void* __restrict__ Xv,
                            const _Float16* __restrict__ Wh,
                            const float* __restrict__ bias,
                            void* __restrict__ Yv,
                            int M, int T) {
    constexpr int K = 512, N = 512;
    __shared__ __align__(16) _Float16 sx[16 * 32];

    const int tid  = threadIdx.x;
    const int lane = tid & 31;
    const int wave = tid >> 5;               // 0..7
    const int r0   = blockIdx.x * 16;
    const int n0   = blockIdx.y * 128 + wave * 16;

    v8f acc = {};
    for (int k0 = 0; k0 < K; k0 += 32) {
#pragma unroll
        for (int e = tid; e < 16 * 32; e += 256) {
            int rr = e >> 5, cc = e & 31;
            if (IN_HALF)
                sx[e] = ((const _Float16*)Xv)[(size_t)(r0 + rr) * K + k0 + cc];
            else
                sx[e] = (_Float16)(((const float*)Xv)[(size_t)(r0 + rr) * K + k0 + cc]);
        }
        __syncthreads();
        v16h a = load_a_frag(sx, 32, lane);
        v16h b = load_b_frag(Wh + (size_t)n0 * K + k0, K, lane);
        acc = WMMA_F16(a, b, acc);
        __syncthreads();
    }

    const int grp = (lane >> 4) & 1;
    const int nn  = lane & 15;
    const int col = n0 + nn;
    const float bv = bias[col];

    if (OUT_TRANS) {
        int h = col >> 6, d = col & 63;       // d_k = 64
        int bb = r0 / T;
        int t0 = (r0 % T) + grp * 8;
        union { _Float16 h8[8]; uint4 u; } pk;
#pragma unroll
        for (int i = 0; i < 8; ++i) pk.h8[i] = (_Float16)(acc[i] + bv);
        *(uint4*)&((_Float16*)Yv)[((size_t)((bb * 8 + h) * 64 + d)) * T + t0] = pk.u;
    } else {
#pragma unroll
        for (int i = 0; i < 8; ++i) {
            float v = acc[i] + bv;
            int row = r0 + i + grp * 8;
            if (OUT_F32) ((float*)Yv)[(size_t)row * N + col] = v;
            else         ((_Float16*)Yv)[(size_t)row * N + col] = (_Float16)v;
        }
    }
}

// ---------------------------------------------------------------------------
// Flash attention, transposed tiles + async double-buffered K/V staging.
// Block: 128 threads = 4 waves sharing (b,h); each wave owns one 16-query
// tile, so the K tile (32x64) and V^T tile (64x32) are staged once per block
// into double-buffered LDS via global_load_async_to_lds_b128 and consumed by
// all 4 waves (ds_load_b128 fragments) -> 4x less global traffic, copy of
// tile t+1 overlaps WMMA work on tile t.
// ---------------------------------------------------------------------------
__device__ __forceinline__ void stage_kv(int tid, lds_half_t* kl, lds_half_t* vl,
                                         const _Float16* kg, const _Float16* vg,
                                         int NF, int T) {
    // K tile: 32 rows (keys) x 64 halfs (d), row stride NF in global, 64 in LDS
#pragma unroll
    for (int c = tid; c < 256; c += 128) {
        int row = c >> 3, c8 = (c & 7) * 8;
        async_copy_b128(kl + row * 64 + c8, kg + (size_t)row * NF + c8);
    }
    // V^T tile: 64 rows (d) x 32 halfs (keys), row stride T in global, 32 in LDS
#pragma unroll
    for (int c = tid; c < 256; c += 128) {
        int row = c >> 2, c8 = (c & 3) * 8;
        async_copy_b128(vl + row * 32 + c8, vg + (size_t)row * T + c8);
    }
}

__global__ void flash_attn_kernel(const _Float16* __restrict__ Qh,
                                  const _Float16* __restrict__ Kh,
                                  const _Float16* __restrict__ Vt,
                                  const int* __restrict__ mask,
                                  _Float16* __restrict__ Oh,
                                  int T) {
    constexpr int NF = 512, DK = 64, H = 8;
    __shared__ __align__(16) _Float16 kbuf[2][32 * 64];
    __shared__ __align__(16) _Float16 vbuf[2][64 * 32];
    __shared__ __align__(16) _Float16 pbuf[4][16 * 32];

    const int tid  = threadIdx.x;
    const int lane = tid & 31;
    const int wave = tid >> 5;                 // 0..3
    const int h    = blockIdx.y;
    const int b    = blockIdx.z;
    const int q0   = (blockIdx.x * 4 + wave) * 16;
    const int grp  = (lane >> 4) & 1;
    const int nn   = lane & 15;

    // Q as B fragments (hoisted): B[k=d][n=q] from row-major Q rows
    const _Float16* qbase = Qh + (size_t)(b * T + q0) * NF + h * DK;
    const v16h bq0 = load_b_frag(qbase, NF, lane);        // d 0..31
    const v16h bq1 = load_b_frag(qbase + 32, NF, lane);   // d 32..63

    v8f acc0 = {}, acc1 = {}, acc2 = {}, acc3 = {};
    float m_run = -1e30f, l_run = 0.0f;        // per-lane (query = nn)

    _Float16* pl = pbuf[wave];
    const float sscale = 0.125f;               // 1/sqrt(d_k), d_k = 64
    const _Float16* kgbase = Kh + (size_t)(b * T) * NF + h * DK;
    const _Float16* vhead  = Vt + (size_t)((b * H + h) * DK) * T;
    const int* mrow_base   = mask + b * T;

    // prologue: stage tile 0
    stage_kv(tid, (lds_half_t*)kbuf[0], (lds_half_t*)vbuf[0], kgbase, vhead, NF, T);
    wait_asynccnt0();
    __syncthreads();

    for (int kt = 0; kt < T; kt += 32) {
        const int cur = (kt >> 5) & 1;
        const _Float16* kc = kbuf[cur];
        const _Float16* vc = vbuf[cur];

        if (kt + 32 < T) {
            stage_kv(tid, (lds_half_t*)kbuf[cur ^ 1], (lds_half_t*)vbuf[cur ^ 1],
                     kgbase + (size_t)(kt + 32) * NF, vhead + kt + 32, NF, T);
            __builtin_prefetch(kgbase + (size_t)(kt + 64) * NF, 0, 1);
            __builtin_prefetch(vhead + kt + 64, 0, 1);
        }

        // K as A fragments from LDS: M = keys, K = d
        v16h ak00 = load_a_frag(kc,                64, lane);  // keys 0..15, d 0..31
        v16h ak01 = load_a_frag(kc + 32,           64, lane);  // keys 0..15, d 32..63
        v16h ak10 = load_a_frag(kc + 16 * 64,      64, lane);  // keys 16..31
        v16h ak11 = load_a_frag(kc + 16 * 64 + 32, 64, lane);

        v8f zero = {};
        v8f s0 = WMMA_F16(ak00, bq0, zero);    // S^T keys kt..+15   x 16 queries
        s0     = WMMA_F16(ak01, bq1, s0);
        v8f s1 = WMMA_F16(ak10, bq0, zero);    // S^T keys +16..+31
        s1     = WMMA_F16(ak11, bq1, s1);

        // mask + scale; element i of s0/s1 -> key kt + grp*8 + i (+16)
        const int* mrow = mrow_base + kt + grp * 8;
        float tmax = -1e30f;
#pragma unroll
        for (int i = 0; i < 8; ++i) {
            float v0 = mrow[i]      ? s0[i] * sscale : -1e30f;
            float v1 = mrow[16 + i] ? s1[i] * sscale : -1e30f;
            s0[i] = v0; s1[i] = v1;
            tmax = fmaxf(tmax, fmaxf(v0, v1));
        }
        tmax = fmaxf(tmax, __shfl_xor(tmax, 16, 32));   // combine both key halves
        float mn = fmaxf(m_run, tmax);
        float sc = __expf(m_run - mn);
        m_run = mn;

        union { _Float16 h8[8]; uint4 u; } pk0, pk1;
        float rs = 0.0f;
#pragma unroll
        for (int i = 0; i < 8; ++i) {
            float p0 = __expf(s0[i] - mn);
            float p1 = __expf(s1[i] - mn);
            rs += p0 + p1;
            pk0.h8[i] = (_Float16)p0;
            pk1.h8[i] = (_Float16)p1;
        }
        rs += __shfl_xor(rs, 16, 32);
        l_run = l_run * sc + rs;
#pragma unroll
        for (int i = 0; i < 8; ++i) {
            acc0[i] *= sc; acc1[i] *= sc; acc2[i] *= sc; acc3[i] *= sc;
        }

        // stage P^T as row-major [q][key] (keys consecutive per lane)
        *(uint4*)&pl[nn * 32 + grp * 8]      = pk0.u;
        *(uint4*)&pl[nn * 32 + 16 + grp * 8] = pk1.u;
        asm volatile("s_wait_dscnt 0" ::: "memory");

        v16h pb  = load_b_frag(pl, 32, lane);            // P^T as B: k=key, n=q
        v16h av0 = load_a_frag(vc,           32, lane);  // V^T d 0..15
        v16h av1 = load_a_frag(vc + 16 * 32, 32, lane);
        v16h av2 = load_a_frag(vc + 32 * 32, 32, lane);
        v16h av3 = load_a_frag(vc + 48 * 32, 32, lane);
        acc0 = WMMA_F16(av0, pb, acc0);                  // O^T tiles
        acc1 = WMMA_F16(av1, pb, acc1);
        acc2 = WMMA_F16(av2, pb, acc2);
        acc3 = WMMA_F16(av3, pb, acc3);

        // own async copies (next buffer) complete + all waves done with cur
        wait_asynccnt0();
        __syncthreads();
    }

    // epilogue: out = acc / (l * (1 + eps)); O^T element i -> d = j*16+grp*8+i,
    // lane -> query row q0+nn. 8 consecutive d per lane => packed b128 stores.
    float inv = 1.0f / (l_run * (1.0f + 1e-8f));
    _Float16* orow = Oh + (size_t)(b * T + q0 + nn) * NF + h * DK + grp * 8;
    v8f* accs[4] = { &acc0, &acc1, &acc2, &acc3 };
#pragma unroll
    for (int j = 0; j < 4; ++j) {
        union { _Float16 h8[8]; uint4 u; } pk;
#pragma unroll
        for (int i = 0; i < 8; ++i) pk.h8[i] = (_Float16)((*accs[j])[i] * inv);
        *(uint4*)&orow[j * 16] = pk.u;
    }
}

// ---------------------------------------------------------------------------
// Host launcher
// ---------------------------------------------------------------------------
extern "C" void kernel_launch(void* const* d_in, const int* in_sizes, int n_in,
                              void* d_out, int out_size, void* d_ws, size_t ws_size,
                              hipStream_t stream) {
    const float* query = (const float*)d_in[0];
    const float* key   = (const float*)d_in[1];
    const float* value = (const float*)d_in[2];
    const int*   mask  = (const int*)d_in[3];
    const float* Wq = (const float*)d_in[4];
    const float* bq = (const float*)d_in[5];
    const float* Wk = (const float*)d_in[6];
    const float* bk = (const float*)d_in[7];
    const float* Wv = (const float*)d_in[8];
    const float* bv = (const float*)d_in[9];
    const float* Wo = (const float*)d_in[10];
    const float* bo = (const float*)d_in[11];

    const int NFEAT = 512;
    const int T = 2048;
    const int M = in_sizes[0] / NFEAT;   // B*T = 8192
    const int B = M / T;                 // 4
    const int WN = NFEAT * NFEAT;        // 262144 weight elements

    // workspace (f16): Qh, Kh, Vt (transposed [B,H,dk,T]), attn out, 4x weights
    _Float16* Qh  = (_Float16*)d_ws;
    _Float16* Kh  = Qh  + (size_t)M * NFEAT;
    _Float16* Vt  = Kh  + (size_t)M * NFEAT;
    _Float16* Oh  = Vt  + (size_t)M * NFEAT;
    _Float16* WqH = Oh  + (size_t)M * NFEAT;
    _Float16* WkH = WqH + (size_t)WN;
    _Float16* WvH = WkH + (size_t)WN;
    _Float16* WoH = WvH + (size_t)WN;

    dim3 cblk(256), cgrd((WN + 255) / 256);
    cvt_f16_kernel<<<cgrd, cblk, 0, stream>>>(Wq, WqH, WN);
    cvt_f16_kernel<<<cgrd, cblk, 0, stream>>>(Wk, WkH, WN);
    cvt_f16_kernel<<<cgrd, cblk, 0, stream>>>(Wv, WvH, WN);
    cvt_f16_kernel<<<cgrd, cblk, 0, stream>>>(Wo, WoH, WN);

    dim3 pblk(256);
    dim3 pgrd(M / 16, NFEAT / 128);
    proj_kernel<false, false, false><<<pgrd, pblk, 0, stream>>>(query, WqH, bq, Qh, M, T);
    proj_kernel<false, false, false><<<pgrd, pblk, 0, stream>>>(key,   WkH, bk, Kh, M, T);
    proj_kernel<false, false, true ><<<pgrd, pblk, 0, stream>>>(value, WvH, bv, Vt, M, T);

    dim3 fgrd(T / 64, 8, B);             // 4 q-tiles of 16 per block
    flash_attn_kernel<<<fgrd, dim3(128), 0, stream>>>(Qh, Kh, Vt, mask, Oh, T);

    proj_kernel<true, true, false><<<pgrd, pblk, 0, stream>>>(Oh, WoH, bo, d_out, M, T);
}
